// MaskedColumnModelingObjective_24455543783503
// MI455X (gfx1250) — matrix operand order repaired and checked by reading, not compiled, not run on hardware
//
#include <hip/hip_runtime.h>
#include <hip/hip_bf16.h>

// Problem dims
#define BB 128
#define NN 512
#define DD 256
#define NT 6
#define NS 8
#define LN_EPS 1e-5f

typedef __bf16 bf16_t;
typedef __attribute__((ext_vector_type(16))) __bf16 v16bf;
typedef __attribute__((ext_vector_type(8)))  __bf16 v8bf;
typedef __attribute__((ext_vector_type(4)))  __bf16 v4bf;
typedef __attribute__((ext_vector_type(8)))  float  v8f;
typedef __attribute__((ext_vector_type(2)))  int    int2v;

typedef __attribute__((address_space(1))) int2v g_int2v;  // global b64 payload
typedef __attribute__((address_space(3))) int2v l_int2v;  // LDS b64 payload

// ---------------------------------------------------------------------------
// gfx1250 async global->LDS path (ASYNCcnt-tracked), with safe fallback.
#if __has_builtin(__builtin_amdgcn_global_load_async_to_lds_b64)
#define USE_ASYNC_LDS 1
#else
#define USE_ASYNC_LDS 0
#endif

__device__ __forceinline__ void async_wait0() {
#if USE_ASYNC_LDS
#if __has_builtin(__builtin_amdgcn_s_wait_asynccnt)
  __builtin_amdgcn_s_wait_asynccnt(0);
#else
  asm volatile("s_wait_asynccnt 0x0" ::: "memory");
#endif
#endif
}

// Stage one 16x32 bf16 A-panel K-slice (1024 B) into LDS: 128 threads x 8 B.
__device__ __forceinline__ void stage_a_async(const bf16_t* __restrict__ aRow0,
                                              bf16_t* lbuf, int k0, int tid) {
  const int row = tid >> 3, chunk = tid & 7;
  const bf16_t* src = aRow0 + (size_t)row * DD + k0 + chunk * 4;
  bf16_t* dst = lbuf + row * 32 + chunk * 4;
#if USE_ASYNC_LDS
  __builtin_amdgcn_global_load_async_to_lds_b64(
      (g_int2v*)(uintptr_t)src, (l_int2v*)(uintptr_t)dst, 0, 0);
#else
  *(double*)dst = *(const double*)src;
#endif
}

// ---------------------------------------------------------------------------
// WMMA fragment loaders (layouts per CDNA5 ISA §7.12.2, wave32)
// A-matrix 16x32 bf16 from LDS: lane<16 holds row M=lane, K chunks
// [0..7],[16..23]; lane>=16 holds [8..15],[24..31].
__device__ __forceinline__ v16bf load_a_frag_lds(const bf16_t* lrow, int half) {
  const bf16_t* q = lrow + (half ? 8 : 0);
  v8bf lo = *reinterpret_cast<const v8bf*>(q);
  v8bf hi = *reinterpret_cast<const v8bf*>(q + 16);
  v16bf a;
#pragma unroll
  for (int i = 0; i < 8; ++i) { a[i] = lo[i]; a[8 + i] = hi[i]; }
  return a;
}
// B-matrix 32x16 bf16: lane<16 holds col N=lane, K chunks [0..7],[16..23];
// lane>=16 holds [8..15],[24..31].
__device__ __forceinline__ v16bf load_b_frag(const bf16_t* __restrict__ colp, int half) {
  const bf16_t* q = colp + (half ? 8 : 0);
  v8bf lo = *reinterpret_cast<const v8bf*>(q);
  v8bf hi = *reinterpret_cast<const v8bf*>(q + 16);
  v16bf b;
#pragma unroll
  for (int i = 0; i < 8; ++i) { b[i] = lo[i]; b[8 + i] = hi[i]; }
  return b;
}

// ---------------------------------------------------------------------------
// Software-pipelined 16xK x Kx64 tile: LDS-staged (async) A, register
// double-buffered B; 4 WMMA tiles per wave, 8 K-steps of 32.
__device__ __forceinline__ void gemm_16x64_pipe(
    const bf16_t* __restrict__ aRow0,   // A panel base (16 rows, stride DD)
    const bf16_t* __restrict__ bCols,   // column c of B = bCols + c*DD (K contig)
    int colbase,                        // wave*64 + (lane&15) [+ tile*16]
    bf16_t* lbuf,                       // __shared__ [2][16*32] bf16
    int tid, int half, int l15, v8f acc[4]) {
  stage_a_async(aRow0, lbuf, 0, tid);
  v16bf bcur[4], bnxt[4];
#pragma unroll
  for (int t = 0; t < 4; ++t)
    bcur[t] = load_b_frag(bCols + (size_t)(colbase + t * 16) * DD, half);
  async_wait0();
  __syncthreads();
#pragma unroll
  for (int kk = 0; kk < 8; ++kk) {
    const int k0 = kk * 32;
    bf16_t* cur = lbuf + (kk & 1) * (16 * 32);
    if (kk < 7) {
      stage_a_async(aRow0, lbuf + ((kk + 1) & 1) * (16 * 32), k0 + 32, tid);
#pragma unroll
      for (int t = 0; t < 4; ++t)
        bnxt[t] = load_b_frag(bCols + (size_t)(colbase + t * 16) * DD + k0 + 32, half);
    }
    v16bf a = load_a_frag_lds(cur + l15 * 32, half);
#pragma unroll
    for (int t = 0; t < 4; ++t)
      acc[t] = __builtin_amdgcn_wmma_f32_16x16x32_bf16(
          false, a, false, bcur[t], (short)0, acc[t], false, false);
    if (kk < 7) {
#pragma unroll
      for (int t = 0; t < 4; ++t) bcur[t] = bnxt[t];
      async_wait0();
      __syncthreads();
    }
  }
}

// ---------------------------------------------------------------------------
__global__ void init_accum_kernel(unsigned* __restrict__ p) {
  if (threadIdx.x < 16) p[threadIdx.x] = 0u;
}

__global__ void convert_x_kernel(const float* __restrict__ x, bf16_t* __restrict__ xb, int n4) {
  int i = blockIdx.x * blockDim.x + threadIdx.x;
  if (i >= n4) return;
  float4 v = reinterpret_cast<const float4*>(x)[i];
  v4bf o;
  o[0] = (bf16_t)v.x; o[1] = (bf16_t)v.y; o[2] = (bf16_t)v.z; o[3] = (bf16_t)v.w;
  reinterpret_cast<v4bf*>(xb)[i] = o;
}

// dst[n*D+k] = (bf16)src[k*D+n]  (transpose so B-fragment K is contiguous)
__global__ void transw_kernel(const float* __restrict__ s0, const float* __restrict__ s1,
                              const float* __restrict__ s2, bf16_t* __restrict__ d0,
                              bf16_t* __restrict__ d1, bf16_t* __restrict__ d2) {
  int idx = blockIdx.x * 256 + threadIdx.x;
  int n = idx / DD, k = idx % DD;
  const float* s = (blockIdx.y == 0) ? s0 : (blockIdx.y == 1) ? s1 : s2;
  bf16_t* d = (blockIdx.y == 0) ? d0 : (blockIdx.y == 1) ? d1 : d2;
  d[(size_t)n * DD + k] = (bf16_t)s[(size_t)k * DD + n];
}

// ---------------------------------------------------------------------------
// Fused MLP head: h = LN(GELU(X@W1 + b1))*g + be ; out = h@W2 + b2 ; masked loss.
__global__ __launch_bounds__(128) void head_kernel(
    const bf16_t* __restrict__ Xbf, const bf16_t* __restrict__ W1t,
    const float* __restrict__ b1, const float* __restrict__ g,
    const float* __restrict__ be, const float* __restrict__ W2,
    const float* __restrict__ b2, const unsigned char* __restrict__ mask,
    const int* __restrict__ tgt, const float* __restrict__ statsT,
    float* __restrict__ fnum, unsigned* __restrict__ ucnt, int isType) {
  __shared__ __align__(16) bf16_t lbuf[2 * 16 * 32];
  __shared__ float hs[16][DD + 4];
  __shared__ float red1[16][8], red2[16][8];
  __shared__ float mu_s[16], rs_s[16];
  __shared__ float logit_s[16][NS];
  __shared__ float blk_sum;
  __shared__ unsigned blk_cnt;

  const int tid = threadIdx.x;
  const int wave = tid >> 5, lane = tid & 31;
  const int half = lane >> 4, l15 = lane & 15;
  const int row0 = blockIdx.x * 16;
  if (tid == 0) { blk_sum = 0.0f; blk_cnt = 0u; }

  v8f acc[4] = {};
  gemm_16x64_pipe(Xbf + (size_t)row0 * DD, W1t, wave * 64 + l15, lbuf,
                  tid, half, l15, acc);

  // bias + exact GELU -> LDS
#pragma unroll
  for (int t = 0; t < 4; ++t) {
#pragma unroll
    for (int r = 0; r < 8; ++r) {
      int row = r + half * 8;
      int col = wave * 64 + t * 16 + l15;
      float v = acc[t][r] + b1[col];
      v = 0.5f * v * (1.0f + erff(v * 0.70710678118654752f));
      hs[row][col] = v;
    }
  }
  __syncthreads();
  // LayerNorm per row (population var)
  {
    int row = tid >> 3, j = tid & 7;
    float s = 0.0f, s2 = 0.0f;
    for (int i = j; i < DD; i += 8) { float v = hs[row][i]; s += v; s2 += v * v; }
    red1[row][j] = s; red2[row][j] = s2;
  }
  __syncthreads();
  if ((tid & 7) == 0) {
    int row = tid >> 3;
    float s = 0.0f, s2 = 0.0f;
#pragma unroll
    for (int j = 0; j < 8; ++j) { s += red1[row][j]; s2 += red2[row][j]; }
    float mu = s * (1.0f / DD);
    float var = s2 * (1.0f / DD) - mu * mu;
    mu_s[row] = mu; rs_s[row] = rsqrtf(var + LN_EPS);
  }
  __syncthreads();
  for (int e = tid; e < 16 * DD; e += 128) {
    int row = e >> 8, col = e & 255;
    hs[row][col] = (hs[row][col] - mu_s[row]) * rs_s[row] * g[col] + be[col];
  }
  __syncthreads();

  if (isType) {
    if (tid < 16 * NT) {
      int row = tid / NT, j = tid % NT;
      float a = b2[j];
      for (int d = 0; d < DD; ++d) a += hs[row][d] * W2[d * NT + j];
      logit_s[row][j] = a;
    }
    __syncthreads();
    if (tid < 16) {
      int grow = row0 + tid;
      if (mask[grow]) {
        float mx = logit_s[tid][0];
#pragma unroll
        for (int j = 1; j < NT; ++j) mx = fmaxf(mx, logit_s[tid][j]);
        float se = 0.0f;
#pragma unroll
        for (int j = 0; j < NT; ++j) se += expf(logit_s[tid][j] - mx);
        float lse = mx + logf(se);
        float nll = lse - logit_s[tid][tgt[grow]];
        atomicAdd(&blk_sum, nll);
        atomicAdd(&blk_cnt, 1u);
      }
    }
  } else {
    int row = tid >> 3, j = tid & 7;
    float a = b2[j];
    for (int d = 0; d < DD; ++d) a += hs[row][d] * W2[d * NS + j];
    int grow = row0 + row;
    if (mask[grow]) {
      float df = a - statsT[(size_t)grow * NS + j];
      atomicAdd(&blk_sum, df * df);
    }
  }
  __syncthreads();
  if (tid == 0) {
    atomicAdd(fnum, blk_sum);
    if (isType) atomicAdd(ucnt, blk_cnt);
  }
}

// ---------------------------------------------------------------------------
// Y = X @ Wc  (bf16 out)
__global__ __launch_bounds__(128) void ygemm_kernel(
    const bf16_t* __restrict__ Xbf, const bf16_t* __restrict__ Wct,
    bf16_t* __restrict__ Ybf) {
  __shared__ __align__(16) bf16_t lbuf[2 * 16 * 32];
  const int tid = threadIdx.x;
  const int wave = tid >> 5, lane = tid & 31;
  const int half = lane >> 4, l15 = lane & 15;
  const int row0 = blockIdx.x * 16;

  v8f acc[4] = {};
  gemm_16x64_pipe(Xbf + (size_t)row0 * DD, Wct, wave * 64 + l15, lbuf,
                  tid, half, l15, acc);
#pragma unroll
  for (int t = 0; t < 4; ++t) {
#pragma unroll
    for (int r = 0; r < 8; ++r) {
      int row = row0 + r + half * 8;
      int col = wave * 64 + t * 16 + l15;
      Ybf[(size_t)row * DD + col] = (bf16_t)acc[t][r];
    }
  }
}

// ---------------------------------------------------------------------------
// Per-batch S = Y @ X^T ; corr = tanh(S+bc), diag=1 ; masked MSE accumulate.
// grid (N/16, N/256, B); B-fragment cols = rows of X (free transpose).
__global__ __launch_bounds__(128) void corr_kernel(
    const bf16_t* __restrict__ Ybf, const bf16_t* __restrict__ Xbf,
    const float* __restrict__ corrT, const unsigned char* __restrict__ mask,
    const float* __restrict__ bcp, float* __restrict__ fnum,
    unsigned* __restrict__ ucnt) {
  __shared__ __align__(16) bf16_t lbuf[2 * 16 * 32];
  __shared__ float blk_sum;
  __shared__ unsigned blk_cnt;
  const int tid = threadIdx.x;
  const int wave = tid >> 5, lane = tid & 31;
  const int half = lane >> 4, l15 = lane & 15;
  const int b = blockIdx.z;
  const int i0 = blockIdx.x * 16;
  const int j0 = blockIdx.y * 256;
  if (tid == 0) { blk_sum = 0.0f; blk_cnt = 0u; }
  __syncthreads();

  v8f acc[4] = {};
  gemm_16x64_pipe(Ybf + (size_t)(b * NN + i0) * DD, Xbf + (size_t)b * NN * DD,
                  j0 + wave * 64 + l15, lbuf, tid, half, l15, acc);

  const float bc = *bcp;
  const float* cb = corrT + (size_t)b * NN * NN;
  const unsigned char* mb = mask + (size_t)b * NN;
  float lsum = 0.0f; unsigned lcnt = 0u;
#pragma unroll
  for (int t = 0; t < 4; ++t) {
#pragma unroll
    for (int r = 0; r < 8; ++r) {
      int i = i0 + r + half * 8;
      int j = j0 + wave * 64 + t * 16 + l15;
      float v = tanhf(acc[t][r] + bc);
      if (i == j) v = 1.0f;
      if (mb[i] | mb[j]) {
        float df = v - cb[(size_t)i * NN + j];
        lsum += df * df;
        ++lcnt;
      }
    }
  }
  atomicAdd(&blk_sum, lsum);
  if (lcnt) atomicAdd(&blk_cnt, lcnt);
  __syncthreads();
  if (tid == 0) {
    atomicAdd(fnum, blk_sum);
    atomicAdd(ucnt, blk_cnt);
  }
}

// ---------------------------------------------------------------------------
__global__ void finalize_kernel(const float* __restrict__ fa,
                                const unsigned* __restrict__ ua,
                                float* __restrict__ out) {
  float cnt = (float)ua[0];
  float ccnt = (float)ua[1];
  float dt = fmaxf(cnt, 1.0f);
  float ds = fmaxf(8.0f * cnt, 1.0f);
  float dc = fmaxf(ccnt, 1.0f);
  out[0] = fa[0] / dt + fa[1] / ds + 0.5f * fa[2] / dc;
}

// ---------------------------------------------------------------------------
extern "C" void kernel_launch(void* const* d_in, const int* in_sizes, int n_in,
                              void* d_out, int out_size, void* d_ws, size_t ws_size,
                              hipStream_t stream) {
  (void)in_sizes; (void)n_in; (void)out_size; (void)ws_size;
  const float* x          = (const float*)d_in[0];
  const unsigned char* m  = (const unsigned char*)d_in[1];
  const int* tgt          = (const int*)d_in[2];
  const float* statsT     = (const float*)d_in[3];
  const float* corrT      = (const float*)d_in[4];
  const float* Wt1 = (const float*)d_in[5];
  const float* bt1 = (const float*)d_in[6];
  const float* gt  = (const float*)d_in[7];
  const float* blt = (const float*)d_in[8];
  const float* Wt2 = (const float*)d_in[9];
  const float* bt2 = (const float*)d_in[10];
  const float* Ws1 = (const float*)d_in[11];
  const float* bs1 = (const float*)d_in[12];
  const float* gs  = (const float*)d_in[13];
  const float* bls = (const float*)d_in[14];
  const float* Ws2 = (const float*)d_in[15];
  const float* bs2 = (const float*)d_in[16];
  const float* Wc  = (const float*)d_in[17];
  const float* bc  = (const float*)d_in[18];

  char* ws = (char*)d_ws;
  float* fa = (float*)ws;                 // fa[0]=type_num fa[1]=stats_num fa[2]=corr_num
  unsigned* ua = (unsigned*)ws + 4;       // ua[0]=mask_cnt  ua[1]=corr_cnt
  bf16_t* Xbf  = (bf16_t*)(ws + 64);
  bf16_t* Ybf  = Xbf + (size_t)BB * NN * DD;
  bf16_t* Wt1t = Ybf + (size_t)BB * NN * DD;
  bf16_t* Ws1t = Wt1t + (size_t)DD * DD;
  bf16_t* Wct  = Ws1t + (size_t)DD * DD;

  init_accum_kernel<<<1, 32, 0, stream>>>((unsigned*)ws);

  const int n4 = BB * NN * DD / 4;
  convert_x_kernel<<<(n4 + 255) / 256, 256, 0, stream>>>(x, Xbf, n4);

  transw_kernel<<<dim3(DD * DD / 256, 3), 256, 0, stream>>>(Wt1, Ws1, Wc, Wt1t, Ws1t, Wct);

  const int nRowBlocks = BB * NN / 16;  // 4096
  head_kernel<<<nRowBlocks, 128, 0, stream>>>(Xbf, Wt1t, bt1, gt, blt, Wt2, bt2,
                                              m, tgt, statsT, &fa[0], &ua[0], 1);
  head_kernel<<<nRowBlocks, 128, 0, stream>>>(Xbf, Ws1t, bs1, gs, bls, Ws2, bs2,
                                              m, tgt, statsT, &fa[1], &ua[0], 0);

  ygemm_kernel<<<nRowBlocks, 128, 0, stream>>>(Xbf, Wct, Ybf);

  corr_kernel<<<dim3(NN / 16, NN / 256, BB), 128, 0, stream>>>(
      Ybf, Xbf, corrT, m, bc, &fa[2], &ua[1]);

  finalize_kernel<<<1, 1, 0, stream>>>(fa, ua, (float*)d_out);
}
